// ZuoErJinYouErChu_26079041421881
// MI455X (gfx1250) — matrix operand order repaired and checked by reading, not compiled, not run on hardware
//
#include <hip/hip_runtime.h>
#include <hip/hip_bf16.h>
#include <math.h>

typedef __bf16 bf16_t;
typedef __attribute__((ext_vector_type(16))) __bf16 v16bf;
typedef __attribute__((ext_vector_type(8)))  __bf16 v8bf;
typedef __attribute__((ext_vector_type(4)))  __bf16 v4bf;
typedef __attribute__((ext_vector_type(8)))  float  v8f;

#define BB 16
#define SS 2048
#define DD 1024
#define MM 32
#define BS (BB * SS)          // 32768 rows
#define CH (BB * MM)          // 512 recurrence channels
#define NCHUNK 32
#define CLEN (SS / NCHUNK)    // 64 steps per chunk

// ---- workspace layout (bytes, 256-aligned) ----
#define XBF_OFF    ((size_t)0)              // 32768*1024 bf16 = 64 MB
#define RW_OFF     ((size_t)67108864)       // 32768*32 f32 = 4 MB
#define WW_OFF     ((size_t)71303168)
#define NM_OFF     ((size_t)75497472)
#define RM_OFF     ((size_t)79691776)
#define WFUSE_OFF  ((size_t)83886080)       // 32*1024 f32
#define BOEFF_OFF  ((size_t)84017152)       // 1024 f32
#define WRP_OFF    ((size_t)84021248)       // packed gate weights, 64 KB each
#define WWP_OFF    ((size_t)84086784)
#define WTP_OFF    ((size_t)84152320)
#define WO0P_OFF   ((size_t)84217856)       // packed Wo[:D], 2 MB
#define WFP_OFF    ((size_t)86315008)       // packed Wfuse, 64 KB
#define SA_OFF     ((size_t)86380544)       // chunk decay A, 16384 f32
#define SB_OFF     ((size_t)86446080)       // chunk offset B, 16384 f32
#define SMIN_OFF   ((size_t)86511616)       // chunk incoming state, 16384 f32

static __device__ __forceinline__ v8f vzero8() {
  v8f z = {0.f, 0.f, 0.f, 0.f, 0.f, 0.f, 0.f, 0.f};
  return z;
}

// A-operand loader: row-major bf16 row pointer already offset by (khalf*8).
// e[0..7]  = K[khalf*8 .. +7],  e[8..15] = K[16 + khalf*8 .. +7]
static __device__ __forceinline__ v16bf load_a16(const bf16_t* p) {
  v8bf c0 = *(const v8bf*)(p);
  v8bf c1 = *(const v8bf*)(p + 16);
  v16bf a;
#pragma unroll
  for (int i = 0; i < 8; ++i) { a[i] = c0[i]; a[i + 8] = c1[i]; }
  return a;
}

static __device__ __forceinline__ v8f wmma_bf16(v16bf a, v16bf b, v8f c) {
  return __builtin_amdgcn_wmma_f32_16x16x32_bf16(false, a, false, b,
                                                 (short)0, c, false, false);
}

// ---------------- prep kernels ----------------

// fp32 -> bf16 cast of x, 4 elements / thread
__global__ __launch_bounds__(256) void k_cvt(const float* __restrict__ x,
                                             bf16_t* __restrict__ xbf, int n4) {
  int i = blockIdx.x * 256 + threadIdx.x;
  if (i >= n4) return;
  float4 v = ((const float4*)x)[i];
  v4bf o;
  o[0] = (bf16_t)v.x; o[1] = (bf16_t)v.y; o[2] = (bf16_t)v.z; o[3] = (bf16_t)v.w;
  ((v4bf*)xbf)[i] = o;
}

// Wfuse = Wp @ Wo[D:], bo_eff = bo + bp @ Wo[D:]
__global__ __launch_bounds__(256) void k_fuse(const float* __restrict__ Wp,
                                              const float* __restrict__ Wo,
                                              const float* __restrict__ bp,
                                              const float* __restrict__ bo,
                                              float* __restrict__ Wfuse,
                                              float* __restrict__ bo_eff) {
  int idx = blockIdx.x * 256 + threadIdx.x;     // 0 .. 32*1024-1
  if (idx >= MM * DD) return;
  int m = idx >> 10, e = idx & (DD - 1);
  const float* wo1 = Wo + (size_t)DD * DD + e;  // Wo[D+d][e] = wo1[d*DD]
  float s = 0.f;
  for (int d = 0; d < DD; ++d) s = fmaf(Wp[m * DD + d], wo1[(size_t)d * DD], s);
  Wfuse[idx] = s;
  if (m == 0) {
    float sb = bo[e];
    for (int d = 0; d < DD; ++d) sb = fmaf(bp[d], wo1[(size_t)d * DD], sb);
    bo_eff[e] = sb;
  }
}

// pack a [K,N] row-major f32 matrix into wave32 WMMA B-operand layout (bf16):
// dst[((kt*NT + nt)*32 + lane)*16 + i] = src[(kt*32 + (lane>>4)*8 + (i&7) + (i>>3)*16)*N + nt*16 + (lane&15)]
__global__ __launch_bounds__(256) void k_pack(const float* __restrict__ src,
                                              bf16_t* __restrict__ dst,
                                              int N, int NT, int total) {
  int idx = blockIdx.x * 256 + threadIdx.x;
  if (idx >= total) return;
  int i    = idx & 15;
  int lane = (idx >> 4) & 31;
  int tile = idx >> 9;
  int nt   = tile % NT;
  int kt   = tile / NT;
  int n = nt * 16 + (lane & 15);
  int k = kt * 32 + (lane >> 4) * 8 + (i & 7) + (i >> 3) * 16;
  dst[idx] = (bf16_t)src[(size_t)k * N + n];
}

// ---------------- gate GEMMs: rw, ww, nm ----------------
// 1 wave per 32-row stripe (2 row tiles); 12 accumulators
// (2 row tiles x 3 gates x 2 column tiles), K=1024. Each B tile reused twice.
__global__ __launch_bounds__(32) void k_gates(const bf16_t* __restrict__ xbf,
                                              const bf16_t* __restrict__ WrP,
                                              const bf16_t* __restrict__ WwP,
                                              const bf16_t* __restrict__ WtP,
                                              const float* __restrict__ br,
                                              const float* __restrict__ bw,
                                              const float* __restrict__ bt,
                                              float* __restrict__ rw,
                                              float* __restrict__ ww,
                                              float* __restrict__ nm) {
  const int lane  = threadIdx.x;
  const int row0  = blockIdx.x * 32;
  const int khalf = lane >> 4;
  const bf16_t* aptr0 = xbf + (size_t)(row0 + (lane & 15)) * DD + khalf * 8;
  const bf16_t* aptr1 = aptr0 + (size_t)16 * DD;

  v8f acc[12];
#pragma unroll
  for (int i = 0; i < 12; ++i) acc[i] = vzero8();

  for (int kt = 0; kt < DD / 32; ++kt) {
    v16bf a0 = load_a16(aptr0 + kt * 32);
    v16bf a1 = load_a16(aptr1 + kt * 32);
#pragma unroll
    for (int nt = 0; nt < 2; ++nt) {
      size_t boff = ((size_t)(kt * 2 + nt) * 32 + lane) * 16;
      v16bf bR = *(const v16bf*)(WrP + boff);
      acc[0 + nt] = wmma_bf16(a0, bR, acc[0 + nt]);
      acc[6 + nt] = wmma_bf16(a1, bR, acc[6 + nt]);
      v16bf bW = *(const v16bf*)(WwP + boff);
      acc[2 + nt] = wmma_bf16(a0, bW, acc[2 + nt]);
      acc[8 + nt] = wmma_bf16(a1, bW, acc[8 + nt]);
      v16bf bT = *(const v16bf*)(WtP + boff);
      acc[4 + nt]  = wmma_bf16(a0, bT, acc[4 + nt]);
      acc[10 + nt] = wmma_bf16(a1, bT, acc[10 + nt]);
    }
  }

#pragma unroll
  for (int rt = 0; rt < 2; ++rt) {
    const int rbase = row0 + rt * 16 + 8 * khalf;
    const int ab = rt * 6;
#pragma unroll
    for (int nt = 0; nt < 2; ++nt) {
      int col = nt * 16 + (lane & 15);
      float cR = br[col], cW = bw[col], cT = bt[col];
#pragma unroll
      for (int v = 0; v < 8; ++v) {
        size_t o = (size_t)(rbase + v) * MM + col;
        float gr = acc[ab + 0 + nt][v] + cR;
        float gw = acc[ab + 2 + nt][v] + cW;
        rw[o] = 1.f / (1.f + __expf(-gr));
        ww[o] = 1.f / (1.f + __expf(-gw));
        nm[o] = acc[ab + 4 + nt][v] + cT;
      }
    }
  }
}

// ---------------- parallel memory recurrence (chunked affine scan) ----------
// Step: m <- (1-w)*m + w*n is affine => per-chunk composition (A, B):
//   m_out = A * m_in + B,  A = prod(1-w),  B = chunk result from m_in = 0.

// pass 1: per (channel, chunk) compute (A, B). 16384 threads.
__global__ __launch_bounds__(512) void k_scan1(const float* __restrict__ ww,
                                               const float* __restrict__ nm,
                                               float* __restrict__ Ac,
                                               float* __restrict__ Bc) {
  const int t = threadIdx.x;      // channel 0..511 (lanes cover m: coalesced)
  const int c = blockIdx.x;       // chunk 0..31
  const size_t base = (size_t)(t >> 5) * SS * MM + (t & 31) + (size_t)c * CLEN * MM;
  float A = 1.f, m = 0.f;
#pragma unroll 4
  for (int s = 0; s < CLEN; ++s) {
    size_t o = base + (size_t)s * MM;
    float w = ww[o], nv = nm[o];
    A *= (1.f - w);
    m = fmaf(w, nv - m, m);
  }
  Ac[c * CH + t] = A;
  Bc[c * CH + t] = m;
}

// pass 2: sequential carry over the 32 chunks, per channel. 512 threads.
__global__ __launch_bounds__(256) void k_scan2(const float* __restrict__ Ac,
                                               const float* __restrict__ Bc,
                                               float* __restrict__ mIn) {
  int t = blockIdx.x * 256 + threadIdx.x;
  if (t >= CH) return;
  float m = 0.f;
#pragma unroll
  for (int c = 0; c < NCHUNK; ++c) {
    mIn[c * CH + t] = m;
    m = fmaf(Ac[c * CH + t], m, Bc[c * CH + t]);
  }
}

// pass 3: replay chunks with correct incoming state, emit rm = rw * m_pre.
__global__ __launch_bounds__(512) void k_scan3(const float* __restrict__ rw,
                                               const float* __restrict__ ww,
                                               const float* __restrict__ nm,
                                               const float* __restrict__ mIn,
                                               float* __restrict__ rm) {
  const int t = threadIdx.x;
  const int c = blockIdx.x;
  const size_t base = (size_t)(t >> 5) * SS * MM + (t & 31) + (size_t)c * CLEN * MM;
  float m = mIn[c * CH + t];
#pragma unroll 4
  for (int s = 0; s < CLEN; ++s) {
    size_t o = base + (size_t)s * MM;
    float r = rw[o], w = ww[o], nv = nm[o];
    rm[o] = r * m;                 // read BEFORE update
    m = fmaf(w, nv - m, m);        // (1-w)*m + w*n
  }
}

// ---------------- fused output GEMM + tanh ----------------
// out = tanh(x @ Wo0 + rm @ Wfuse + bo_eff)
// block = 8 waves; wave w computes column tiles [w*8, w*8+8) for a 32-row
// stripe (2 row tiles) => each B tile load feeds 2 WMMAs.
__global__ __launch_bounds__(256) void k_out(const bf16_t* __restrict__ xbf,
                                             const float* __restrict__ rm,
                                             const bf16_t* __restrict__ Wo0P,
                                             const bf16_t* __restrict__ WfP,
                                             const float* __restrict__ bo_eff,
                                             float* __restrict__ out) {
  const int lane  = threadIdx.x & 31;
  const int wave  = threadIdx.x >> 5;       // 0..7
  const int row0  = blockIdx.x * 32;
  const int khalf = lane >> 4;
  const int ct0   = wave * 8;               // first column tile
  const bf16_t* aptr0 = xbf + (size_t)(row0 + (lane & 15)) * DD + khalf * 8;
  const bf16_t* aptr1 = aptr0 + (size_t)16 * DD;

  v8f acc[16];                              // [row tile][col tile j]
#pragma unroll
  for (int j = 0; j < 16; ++j) acc[j] = vzero8();

  for (int kt = 0; kt < DD / 32; ++kt) {
    v16bf a0 = load_a16(aptr0 + kt * 32);
    v16bf a1 = load_a16(aptr1 + kt * 32);
#pragma unroll
    for (int j = 0; j < 8; ++j) {
      size_t boff = ((size_t)(kt * (DD / 16) + ct0 + j) * 32 + lane) * 16;
      v16bf b = *(const v16bf*)(Wo0P + boff);
      acc[j]     = wmma_bf16(a0, b, acc[j]);
      acc[8 + j] = wmma_bf16(a1, b, acc[8 + j]);
    }
  }

  // rm @ Wfuse : single K=32 step, cast rm tiles to bf16 on the fly
  const float* rptr0 = rm + (size_t)(row0 + (lane & 15)) * MM + khalf * 8;
  const float* rptr1 = rptr0 + (size_t)16 * MM;
  v16bf ar0, ar1;
#pragma unroll
  for (int i = 0; i < 8; ++i) {
    ar0[i]     = (bf16_t)rptr0[i];
    ar0[i + 8] = (bf16_t)rptr0[16 + i];
    ar1[i]     = (bf16_t)rptr1[i];
    ar1[i + 8] = (bf16_t)rptr1[16 + i];
  }
#pragma unroll
  for (int j = 0; j < 8; ++j) {
    size_t boff = ((size_t)(ct0 + j) * 32 + lane) * 16;
    v16bf b = *(const v16bf*)(WfP + boff);
    acc[j]     = wmma_bf16(ar0, b, acc[j]);
    acc[8 + j] = wmma_bf16(ar1, b, acc[8 + j]);
  }

#pragma unroll
  for (int rt = 0; rt < 2; ++rt) {
    const int rbase = row0 + rt * 16 + 8 * khalf;
#pragma unroll
    for (int j = 0; j < 8; ++j) {
      int col = (ct0 + j) * 16 + (lane & 15);
      float bb = bo_eff[col];
#pragma unroll
      for (int v = 0; v < 8; ++v)
        out[(size_t)(rbase + v) * DD + col] = tanhf(acc[rt * 8 + j][v] + bb);
    }
  }
}

// ---------------- host launcher ----------------
extern "C" void kernel_launch(void* const* d_in, const int* in_sizes, int n_in,
                              void* d_out, int out_size, void* d_ws, size_t ws_size,
                              hipStream_t stream) {
  const float* x  = (const float*)d_in[0];
  const float* Wr = (const float*)d_in[1];
  const float* br = (const float*)d_in[2];
  const float* Ww = (const float*)d_in[3];
  const float* bw = (const float*)d_in[4];
  const float* Wt = (const float*)d_in[5];
  const float* bt = (const float*)d_in[6];
  const float* Wp = (const float*)d_in[7];
  const float* bp = (const float*)d_in[8];
  const float* Wo = (const float*)d_in[9];
  const float* bo = (const float*)d_in[10];
  float* out = (float*)d_out;

  char* ws = (char*)d_ws;
  bf16_t* xbf    = (bf16_t*)(ws + XBF_OFF);
  float*  rw     = (float*)(ws + RW_OFF);
  float*  wwg    = (float*)(ws + WW_OFF);
  float*  nm     = (float*)(ws + NM_OFF);
  float*  rm     = (float*)(ws + RM_OFF);
  float*  Wfuse  = (float*)(ws + WFUSE_OFF);
  float*  bo_eff = (float*)(ws + BOEFF_OFF);
  bf16_t* WrP    = (bf16_t*)(ws + WRP_OFF);
  bf16_t* WwP    = (bf16_t*)(ws + WWP_OFF);
  bf16_t* WtP    = (bf16_t*)(ws + WTP_OFF);
  bf16_t* Wo0P   = (bf16_t*)(ws + WO0P_OFF);
  bf16_t* WfP    = (bf16_t*)(ws + WFP_OFF);
  float*  Ac     = (float*)(ws + SA_OFF);
  float*  Bc     = (float*)(ws + SB_OFF);
  float*  mIn    = (float*)(ws + SMIN_OFF);

  // 1) x -> bf16
  {
    int n4 = (BS * DD) / 4;
    k_cvt<<<(n4 + 255) / 256, 256, 0, stream>>>(x, xbf, n4);
  }
  // 2) Wfuse = Wp @ Wo[D:], bo_eff
  k_fuse<<<(MM * DD + 255) / 256, 256, 0, stream>>>(Wp, Wo, bp, bo, Wfuse, bo_eff);
  // 3) pack weights into WMMA B layouts
  k_pack<<<(DD * MM + 255) / 256, 256, 0, stream>>>(Wr, WrP, MM, MM / 16, DD * MM);
  k_pack<<<(DD * MM + 255) / 256, 256, 0, stream>>>(Ww, WwP, MM, MM / 16, DD * MM);
  k_pack<<<(DD * MM + 255) / 256, 256, 0, stream>>>(Wt, WtP, MM, MM / 16, DD * MM);
  k_pack<<<(DD * DD + 255) / 256, 256, 0, stream>>>(Wo, Wo0P, DD, DD / 16, DD * DD);
  k_pack<<<(MM * DD + 255) / 256, 256, 0, stream>>>(Wfuse, WfP, DD, DD / 16, MM * DD);
  // 4) gates
  k_gates<<<BS / 32, 32, 0, stream>>>(xbf, WrP, WwP, WtP, br, bw, bt, rw, wwg, nm);
  // 5) recurrence: chunked affine scan
  k_scan1<<<NCHUNK, 512, 0, stream>>>(wwg, nm, Ac, Bc);
  k_scan2<<<2, 256, 0, stream>>>(Ac, Bc, mIn);
  k_scan3<<<NCHUNK, 512, 0, stream>>>(rw, wwg, nm, mIn, rm);
  // 6) fused output GEMM + tanh
  k_out<<<BS / 32, 256, 0, stream>>>(xbf, rm, Wo0P, WfP, bo_eff, out);
}
